// TransformerBlock_90486370992734
// MI455X (gfx1250) — compile-verified, matrix operand order checked by reading, not run on hardware
//
#include <hip/hip_runtime.h>
#include <hip/hip_bf16.h>

typedef __attribute__((ext_vector_type(16))) _Float16 v16h;
typedef __attribute__((ext_vector_type(8)))  float    v8f;
typedef __attribute__((ext_vector_type(8)))  _Float16 h8;
typedef __attribute__((ext_vector_type(4)))  unsigned int u32x4;
typedef __attribute__((ext_vector_type(8)))  int i32x8;
typedef __attribute__((ext_vector_type(4)))  int i32x4;

// ---------------- constants ----------------
#define TOKENS   (2 * 256 * 256)   // B*H*W
#define CDIM     256
#define HEADS    8
#define HDIM     32
#define WN       64                // window tokens (8x8)
#define NWIN     2048              // B * 32 * 32 windows
#define ATT_SCALE 0.17677669529663689f   // (DIM/HEADS)^-0.5
#define BN_SCL   0.99999500003749973f    // 1/sqrt(1+1e-5)

#if defined(__has_builtin)
#  if __has_builtin(__builtin_amdgcn_tensor_load_to_lds)
#    define HAVE_TDM 1
#  endif
#endif
#ifndef HAVE_TDM
#  define HAVE_TDM 0
#endif

#if HAVE_TDM
__device__ __forceinline__ void tdm_load_tile_2d(unsigned ldsOff, const void* gptr,
                                                 int tile0, int tile1, int stride0) {
    // D# group 0: count=1 | lds_addr | global_addr | type=2
    unsigned long long ga = (unsigned long long)(size_t)gptr;
    u32x4 g0 = {};
    g0[0] = 1u;                                            // count=1 (valid descriptor)
    g0[1] = ldsOff;                                        // lds_addr (bytes)
    g0[2] = (unsigned)(ga & 0xFFFFFFFFu);                  // global_addr[31:0]
    g0[3] = (unsigned)((ga >> 32) & 0x01FFFFFFu)           // global_addr[56:32]
          | (2u << 30);                                    // type=2 ("image")
    // D# group 1: data_size=2B, tensor/tile dims, dim0 stride
    i32x8 g1 = {};
    g1[0] = (1 << 16);                                     // data_size=1 -> 2 bytes
    g1[1] = (int)((unsigned)(stride0 & 0xFFFF) << 16);     // tensor_dim0[15:0] (>= tile0)
    g1[2] = (int)(((unsigned)stride0 >> 16) & 0xFFFFu);    // tensor_dim0[31:16]; tensor_dim1 lo16=0
    g1[3] = 16 | (tile1 == 0 ? 0 : 0) | (32 << 16);        // tensor_dim1 hi16 (=1M), tile_dim0=32
    g1[3] = 16 | ((unsigned)tile0 << 16);                  // tensor_dim1=1<<20, tile_dim0
    g1[4] = tile1;                                         // tile_dim1 ; tile_dim2=0
    g1[5] = stride0;                                       // tensor_dim0_stride low32 (elements)
    g1[6] = 0;                                             // stride hi / dim1_stride lo
    g1[7] = 0;
    i32x4 z4 = {};
#if __has_include(<hip/amd_detail/amd_gfx1250_TDM.h>)
    i32x8 z8 = {};
    __builtin_amdgcn_tensor_load_to_lds(g0, g1, z4, z4, z8, 0);
#else
    __builtin_amdgcn_tensor_load_to_lds(g0, g1, z4, z4, 0);
#endif
}
#endif

// ---------------- helpers ----------------
__device__ __forceinline__ int tok_of(int b, int wh, int ww, int i) {
    // shifted-window element i -> original token index (roll folded in)
    int r = i >> 3, c = i & 7;
    int ho = (wh * 8 + r + 4) & 255;
    int wo = (ww * 8 + c + 4) & 255;
    return (b << 16) + (ho << 8) + wo;
}

__device__ __forceinline__ int reg_of(int wh, int ww, int i) {
    int r = i >> 3, c = i & 7;
    int h = wh * 8 + r, w = ww * 8 + c;
    int rh = (h < 248) ? 0 : ((h < 252) ? 1 : 2);
    int rw = (w < 248) ? 0 : ((w < 252) ? 1 : 2);
    return rh * 3 + rw;
}

// ---------------- elementwise kernels ----------------
// Pack f32 weight W[K,N] into WMMA B-fragment order:
// tile (kt,nt) of 32x16, per lane 16 contiguous halves in the ISA lane layout.
__global__ void k_pack_w(const float* __restrict__ src, _Float16* __restrict__ dst,
                         int K, int N) {
    int idx = blockIdx.x * blockDim.x + threadIdx.x;
    int total = (K >> 5) * (N >> 4) * 32;
    if (idx >= total) return;
    int lane = idx & 31;
    int t = idx >> 5;
    int ntiles = N >> 4;
    int nt = t % ntiles, kt = t / ntiles;
    int g = lane >> 4, n = lane & 15;
    int col = nt * 16 + n;
    int kbase = kt * 32 + g * 8;
    _Float16* d = dst + (size_t)idx * 16;
    #pragma unroll
    for (int e = 0; e < 8; e++) {
        d[e]     = (_Float16)src[(size_t)(kbase + e) * N + col];
        d[8 + e] = (_Float16)src[(size_t)(kbase + 16 + e) * N + col];
    }
}

__global__ void k_layernorm(const float* __restrict__ x, const float* __restrict__ g,
                            const float* __restrict__ b, _Float16* __restrict__ out) {
    __shared__ float s1[256], s2[256];
    int t = blockIdx.x, c = threadIdx.x;
    float v = x[(size_t)t * CDIM + c];
    s1[c] = v; s2[c] = v * v;
    __syncthreads();
    for (int s = 128; s > 0; s >>= 1) {
        if (c < s) { s1[c] += s1[c + s]; s2[c] += s2[c + s]; }
        __syncthreads();
    }
    float mean = s1[0] * (1.f / 256.f);
    float var  = s2[0] * (1.f / 256.f) - mean * mean;
    float inv  = rsqrtf(var + 1e-5f);
    out[(size_t)t * CDIM + c] = (_Float16)((v - mean) * inv * g[c] + b[c]);
}

__global__ void k_dwconv(const _Float16* __restrict__ in, const float* __restrict__ wgt,
                         const float* __restrict__ bg, const float* __restrict__ bb,
                         _Float16* __restrict__ out, int C, int dil) {
    long idx = (long)blockIdx.x * blockDim.x + threadIdx.x;
    long total = (long)TOKENS * C;
    if (idx >= total) return;
    int c = (int)(idx % C);
    long t = idx / C;
    int w = (int)(t & 255);
    int h = (int)((t >> 8) & 255);
    int b = (int)(t >> 16);
    float acc = 0.f;
    #pragma unroll
    for (int ky = 0; ky < 3; ky++) {
        int hh = h + (ky - 1) * dil;
        if (hh < 0 || hh >= 256) continue;
        #pragma unroll
        for (int kx = 0; kx < 3; kx++) {
            int ww = w + (kx - 1) * dil;
            if (ww < 0 || ww >= 256) continue;
            size_t src = (((size_t)b << 16) + ((size_t)hh << 8) + ww) * C + c;
            acc += (float)in[src] * wgt[(ky * 3 + kx) * C + c];
        }
    }
    float y = acc * (bg[c] * BN_SCL) + bb[c];
    out[idx] = (_Float16)fmaxf(y, 0.f);
}

__global__ void k_copy_ch(const _Float16* __restrict__ src, _Float16* __restrict__ dst,
                          int Cin, int Cout, int coff) {
    long idx = (long)blockIdx.x * blockDim.x + threadIdx.x;   // one h8 per thread
    long total = (long)TOKENS * (Cin >> 3);
    if (idx >= total) return;
    int c8 = (int)(idx % (Cin >> 3));
    long t = idx / (Cin >> 3);
    *(h8*)(dst + t * (size_t)Cout + coff + c8 * 8) =
        *(const h8*)(src + t * (size_t)Cin + c8 * 8);
}

// ---------------- generic WMMA GEMM ----------------
// C[M,N] = act(A[M,K] @ Wpacked[K,N] * scale + bias) (+ res)
// block: 8 waves; block tile 128(M) x 64(N); A tile staged in LDS (TDM if available)
template <int ACT, bool OUT16>
__global__ __launch_bounds__(256) void k_gemm(
    const _Float16* __restrict__ A, const _Float16* __restrict__ Bp,
    const float* __restrict__ scale, const float* __restrict__ bias,
    const float* __restrict__ res, _Float16* __restrict__ o16,
    float* __restrict__ o32, int N, int K) {
    __shared__ _Float16 Atile[128 * 32];
    int lane = threadIdx.x & 31;
    int wave = threadIdx.x >> 5;
    int rowBase = blockIdx.y * 128 + wave * 16;
    int colBase = blockIdx.x * 64;
    int g = lane >> 4, mn = lane & 15;
    int ntiles = N >> 4;

    v8f acc0 = {}, acc1 = {}, acc2 = {}, acc3 = {};
    for (int k0 = 0; k0 < K; k0 += 32) {
        // ---- stage A block tile (128 x 32) into LDS ----
#if HAVE_TDM
        if (wave == 0) {
            const _Float16* gsrc = A + (size_t)blockIdx.y * 128 * K + k0;
            tdm_load_tile_2d((unsigned)(size_t)&Atile[0], gsrc, 32, 128, K);
            __builtin_amdgcn_s_wait_tensorcnt(0);
        }
        __syncthreads();
#else
        for (int i = threadIdx.x; i < 128 * 32 / 8; i += 256) {
            int row = (i * 8) >> 5, ck = (i * 8) & 31;
            *(h8*)&Atile[row * 32 + ck] =
                *(const h8*)(A + (size_t)(blockIdx.y * 128 + row) * K + k0 + ck);
        }
        __syncthreads();
#endif
        // ---- compute ----
        v16h a;
        {
            const _Float16* ap = &Atile[(wave * 16 + mn) * 32 + g * 8];
            #pragma unroll
            for (int e = 0; e < 8; e++) { a[e] = ap[e]; a[8 + e] = ap[16 + e]; }
        }
        #pragma unroll
        for (int tn = 0; tn < 4; tn++) {
            v16h bf;
            const _Float16* bp = Bp +
                (((size_t)(k0 >> 5) * ntiles + ((colBase >> 4) + tn)) * 32 + lane) * 16;
            #pragma unroll
            for (int e = 0; e < 16; e++) bf[e] = bp[e];
            v8f& acc = (tn == 0) ? acc0 : (tn == 1) ? acc1 : (tn == 2) ? acc2 : acc3;
            acc = __builtin_amdgcn_wmma_f32_16x16x32_f16(false, a, false, bf, (short)0,
                                                         acc, false, false);
        }
        __syncthreads();
    }
    #pragma unroll
    for (int tn = 0; tn < 4; tn++) {
        int col = colBase + tn * 16 + mn;
        float sc = scale ? scale[col] * BN_SCL : 1.f;
        float bs = bias ? bias[col] : 0.f;
        v8f acc = (tn == 0) ? acc0 : (tn == 1) ? acc1 : (tn == 2) ? acc2 : acc3;
        #pragma unroll
        for (int r = 0; r < 8; r++) {
            int row = rowBase + g * 8 + r;
            float v = acc[r] * sc + bs;
            if (ACT == 1) v = fmaxf(v, 0.f);
            if (ACT == 2) v = 0.5f * v * (1.f + erff(v * 0.70710678118654752f));
            if (res) v += res[(size_t)row * N + col];
            if (OUT16) o16[(size_t)row * N + col] = (_Float16)v;
            else       o32[(size_t)row * N + col] = v;
        }
    }
}

// ---------------- windowed attention (one wave per window*head) ----------------
__global__ __launch_bounds__(32) void k_attn(
    const _Float16* __restrict__ q, const _Float16* __restrict__ kv,
    const float* __restrict__ rpb, _Float16* __restrict__ out) {
    __shared__ float    S[WN][WN];
    __shared__ _Float16 P[WN][WN];
    __shared__ _Float16 Vs[WN][HDIM];
    int lane = threadIdx.x;
    int wid  = blockIdx.x >> 3;
    int head = blockIdx.x & 7;
    int b  = wid >> 10;
    int w  = wid & 1023;
    int wh = w >> 5, ww = w & 31;
    int g = lane >> 4, mn = lane & 15;

    // stage V rows (contiguous 32-half reads per row) into LDS
    for (int rr = 0; rr < 2; rr++) {
        int i = lane * 2 + rr;
        const _Float16* vp = kv + (size_t)tok_of(b, wh, ww, i) * 512 + 256 + head * HDIM;
        *(h8*)&Vs[i][0]  = *(const h8*)(vp);
        *(h8*)&Vs[i][8]  = *(const h8*)(vp + 8);
        *(h8*)&Vs[i][16] = *(const h8*)(vp + 16);
        *(h8*)&Vs[i][24] = *(const h8*)(vp + 24);
    }

    // S = Q @ K^T  (hd = 32 => single K step); 4x4 tiles of 16x16
    for (int tm = 0; tm < 4; tm++) {
        v16h a;
        {
            int i = tm * 16 + mn;
            const _Float16* ap = q + (size_t)tok_of(b, wh, ww, i) * CDIM + head * HDIM + g * 8;
            #pragma unroll
            for (int e = 0; e < 8; e++) { a[e] = ap[e]; a[8 + e] = ap[16 + e]; }
        }
        for (int tn = 0; tn < 4; tn++) {
            v16h bf;
            int j = tn * 16 + mn;
            const _Float16* bp = kv + (size_t)tok_of(b, wh, ww, j) * 512 + head * HDIM + g * 8;
            #pragma unroll
            for (int e = 0; e < 8; e++) { bf[e] = bp[e]; bf[8 + e] = bp[16 + e]; }
            v8f c = {};
            c = __builtin_amdgcn_wmma_f32_16x16x32_f16(false, a, false, bf, (short)0,
                                                       c, false, false);
            #pragma unroll
            for (int r = 0; r < 8; r++) S[tm * 16 + g * 8 + r][tn * 16 + mn] = c[r];
        }
    }
    __syncthreads();

    // softmax over 64 cols, 2 rows per lane; bias = rpb gather + shift mask
    for (int rr = 0; rr < 2; rr++) {
        int i  = lane * 2 + rr;
        int ri = i >> 3, ci = i & 7;
        int regi = reg_of(wh, ww, i);
        float mx = -1e30f;
        for (int j = 0; j < 64; j++) {
            int rj = j >> 3, cj = j & 7;
            int rel = (ri - rj + 7) * 15 + (ci - cj + 7);
            float m = (reg_of(wh, ww, j) != regi) ? -100.f : 0.f;
            float v = S[i][j] * ATT_SCALE + rpb[rel * HEADS + head] + m;
            S[i][j] = v;
            mx = fmaxf(mx, v);
        }
        float sum = 0.f;
        for (int j = 0; j < 64; j++) {
            float e = __expf(S[i][j] - mx);
            S[i][j] = e;
            sum += e;
        }
        float inv = 1.f / sum;
        for (int j = 0; j < 64; j++) P[i][j] = (_Float16)(S[i][j] * inv);
    }
    __syncthreads();

    // O = P @ V : 4x2 tiles, K = 64 (two steps); V from LDS
    for (int tm = 0; tm < 4; tm++) {
        for (int tn = 0; tn < 2; tn++) {
            v8f c = {};
            for (int k0 = 0; k0 < 64; k0 += 32) {
                v16h a;
                const _Float16* ap = &P[tm * 16 + mn][k0 + g * 8];
                #pragma unroll
                for (int e = 0; e < 8; e++) { a[e] = ap[e]; a[8 + e] = ap[16 + e]; }
                v16h bf;
                #pragma unroll
                for (int e = 0; e < 8; e++) {
                    bf[e]     = Vs[k0 + g * 8 + e][tn * 16 + mn];
                    bf[8 + e] = Vs[k0 + 16 + g * 8 + e][tn * 16 + mn];
                }
                c = __builtin_amdgcn_wmma_f32_16x16x32_f16(false, a, false, bf, (short)0,
                                                           c, false, false);
            }
            #pragma unroll
            for (int r = 0; r < 8; r++) {
                int i = tm * 16 + g * 8 + r;
                out[(size_t)tok_of(b, wh, ww, i) * CDIM + head * HDIM + tn * 16 + mn] = (_Float16)c[r];
            }
        }
    }
}

// ---------------- host launcher ----------------
extern "C" void kernel_launch(void* const* d_in, const int* in_sizes, int n_in,
                              void* d_out, int out_size, void* d_ws, size_t ws_size,
                              hipStream_t stream) {
    const float* x    = (const float*)d_in[0];
    const float* n1g  = (const float*)d_in[1];
    const float* n1b  = (const float*)d_in[2];
    const float* n2g  = (const float*)d_in[3];
    const float* n2b  = (const float*)d_in[4];
    const float* rpb  = (const float*)d_in[5];
    const float* wq   = (const float*)d_in[6];
    const float* bq   = (const float*)d_in[7];
    const float* wkv  = (const float*)d_in[8];
    const float* bkv  = (const float*)d_in[9];
    const float* wp   = (const float*)d_in[10];
    const float* bp   = (const float*)d_in[11];
    const float* w1   = (const float*)d_in[12];
    const float* b1   = (const float*)d_in[13];
    const float* w2   = (const float*)d_in[14];
    const float* b2   = (const float*)d_in[15];
    const float* cmp  = (const float*)d_in[16];
    const float* s1dw = (const float*)d_in[17];
    const float* s1g1 = (const float*)d_in[18];
    const float* s1b1 = (const float*)d_in[19];
    const float* s1pw = (const float*)d_in[20];
    const float* s1g2 = (const float*)d_in[21];
    const float* s1b2 = (const float*)d_in[22];
    const float* s2dw = (const float*)d_in[23];
    const float* s2g1 = (const float*)d_in[24];
    const float* s2b1 = (const float*)d_in[25];
    const float* s2pw = (const float*)d_in[26];
    const float* s2g2 = (const float*)d_in[27];
    const float* s2b2 = (const float*)d_in[28];
    const float* s3dw = (const float*)d_in[29];
    const float* s3g1 = (const float*)d_in[30];
    const float* s3b1 = (const float*)d_in[31];
    const float* s3pw = (const float*)d_in[32];
    const float* s3g2 = (const float*)d_in[33];
    const float* s3b2 = (const float*)d_in[34];
    const float* wup  = (const float*)d_in[35];
    const float* ng   = (const float*)d_in[36];
    const float* nb   = (const float*)d_in[37];
    float* outp = (float*)d_out;

    char* ws = (char*)d_ws;
    size_t off = 0;
    auto alloc = [&](size_t bytes) -> void* {
        void* p = ws + off;
        off += (bytes + 255) & ~(size_t)255;
        return p;
    };
    const size_t T = TOKENS;
    _Float16* wq16  = (_Float16*)alloc(256 * 256 * 2);
    _Float16* wkv16 = (_Float16*)alloc(256 * 512 * 2);
    _Float16* wp16  = (_Float16*)alloc(256 * 256 * 2);
    _Float16* w1_16 = (_Float16*)alloc(256 * 1024 * 2);
    _Float16* w2_16 = (_Float16*)alloc(1024 * 256 * 2);
    _Float16* cmp16 = (_Float16*)alloc(256 * 128 * 2);
    _Float16* pw1_16 = (_Float16*)alloc(128 * 128 * 2);
    _Float16* pw2_16 = (_Float16*)alloc(256 * 128 * 2);
    _Float16* pw3_16 = (_Float16*)alloc(384 * 128 * 2);
    _Float16* up16  = (_Float16*)alloc(512 * 256 * 2);
    _Float16* xn    = (_Float16*)alloc(T * 256 * 2);
    _Float16* qb    = (_Float16*)alloc(T * 256 * 2);
    _Float16* kvb   = (_Float16*)alloc(T * 512 * 2);
    _Float16* x0    = (_Float16*)alloc(T * 128 * 2);
    _Float16* t1    = (_Float16*)alloc(T * 128 * 2);
    _Float16* x1    = (_Float16*)alloc(T * 128 * 2);
    _Float16* cat2  = (_Float16*)alloc(T * 256 * 2);
    _Float16* t2    = (_Float16*)alloc(T * 256 * 2);
    _Float16* x2l   = (_Float16*)alloc(T * 128 * 2);
    _Float16* cat3  = (_Float16*)alloc(T * 384 * 2);
    _Float16* t3    = (_Float16*)alloc(T * 384 * 2);
    _Float16* x3    = (_Float16*)alloc(T * 128 * 2);
    _Float16* cat4  = (_Float16*)alloc(T * 512 * 2);
    _Float16* locb  = (_Float16*)alloc(T * 256 * 2);
    _Float16* atno  = (_Float16*)alloc(T * 256 * 2);
    float*    x2res = (float*)alloc(T * 256 * 4);
    _Float16* xn2   = (_Float16*)alloc(T * 256 * 2);
    _Float16* h1    = (_Float16*)alloc(T * 1024 * 2);

    auto pack = [&](const float* s, _Float16* d, int K, int N) {
        int total = (K >> 5) * (N >> 4) * 32;
        k_pack_w<<<(total + 255) / 256, 256, 0, stream>>>(s, d, K, N);
    };
    pack(wq,  wq16,  256, 256);  pack(wkv, wkv16, 256, 512);
    pack(wp,  wp16,  256, 256);  pack(w1,  w1_16, 256, 1024);
    pack(w2,  w2_16, 1024, 256); pack(cmp, cmp16, 256, 128);
    pack(s1pw, pw1_16, 128, 128); pack(s2pw, pw2_16, 256, 128);
    pack(s3pw, pw3_16, 384, 128); pack(wup, up16, 512, 256);

    auto gemm = [&](const _Float16* A, const _Float16* Bm, const float* scale,
                    const float* bias, const float* res, _Float16* o16, float* o32,
                    int Nn, int K, int act) {
        dim3 grid(Nn / 64, (unsigned)(T / 128));
        if (o16) {
            if (act == 0)      k_gemm<0, true><<<grid, 256, 0, stream>>>(A, Bm, scale, bias, res, o16, nullptr, Nn, K);
            else if (act == 1) k_gemm<1, true><<<grid, 256, 0, stream>>>(A, Bm, scale, bias, res, o16, nullptr, Nn, K);
            else               k_gemm<2, true><<<grid, 256, 0, stream>>>(A, Bm, scale, bias, res, o16, nullptr, Nn, K);
        } else {
            k_gemm<0, false><<<grid, 256, 0, stream>>>(A, Bm, scale, bias, res, nullptr, o32, Nn, K);
        }
    };
    auto dwc = [&](const _Float16* in, const float* wgt, const float* bg,
                   const float* bb, _Float16* o, int C, int dil) {
        long total = (long)T * C;
        k_dwconv<<<(unsigned)((total + 255) / 256), 256, 0, stream>>>(in, wgt, bg, bb, o, C, dil);
    };
    auto cpy = [&](const _Float16* s, _Float16* d, int Cin, int Cout, int coff) {
        long total = (long)T * (Cin >> 3);
        k_copy_ch<<<(unsigned)((total + 255) / 256), 256, 0, stream>>>(s, d, Cin, Cout, coff);
    };

    // LN1
    k_layernorm<<<(unsigned)T, 256, 0, stream>>>(x, n1g, n1b, xn);
    // q projection
    gemm(xn, wq16, nullptr, bq, nullptr, qb, nullptr, 256, 256, 0);
    // local branch
    gemm(xn, cmp16, nullptr, nullptr, nullptr, x0, nullptr, 128, 256, 0);
    dwc(x0, s1dw, s1g1, s1b1, t1, 128, 1);
    gemm(t1, pw1_16, s1g2, s1b2, nullptr, x1, nullptr, 128, 128, 1);
    cpy(x1, cat2, 128, 256, 0); cpy(x0, cat2, 128, 256, 128);
    dwc(cat2, s2dw, s2g1, s2b1, t2, 256, 2);
    gemm(t2, pw2_16, s2g2, s2b2, nullptr, x2l, nullptr, 128, 256, 1);
    cpy(x2l, cat3, 128, 384, 0); cpy(x1, cat3, 128, 384, 128); cpy(x0, cat3, 128, 384, 256);
    dwc(cat3, s3dw, s3g1, s3b1, t3, 384, 3);
    gemm(t3, pw3_16, s3g2, s3b2, nullptr, x3, nullptr, 128, 384, 1);
    cpy(x3, cat4, 128, 512, 0); cpy(x2l, cat4, 128, 512, 128);
    cpy(x1, cat4, 128, 512, 256); cpy(x0, cat4, 128, 512, 384);
    gemm(cat4, up16, ng, nb, nullptr, locb, nullptr, 256, 512, 0);
    // kv projection from local branch
    gemm(locb, wkv16, nullptr, bkv, nullptr, kvb, nullptr, 512, 256, 0);
    // shifted-window attention
    k_attn<<<NWIN * HEADS, 32, 0, stream>>>(qb, kvb, rpb, atno);
    // output proj + residual (f32)
    gemm(atno, wp16, nullptr, bp, x, nullptr, x2res, 256, 256, 0);
    // LN2 + MLP
    k_layernorm<<<(unsigned)T, 256, 0, stream>>>(x2res, n2g, n2b, xn2);
    gemm(xn2, w1_16, nullptr, b1, nullptr, h1, nullptr, 1024, 256, 2);
    gemm(h1, w2_16, nullptr, b2, x2res, nullptr, outp, 256, 1024, 0);
}
